// locationAttention_3856880632208
// MI455X (gfx1250) — matrix operand order repaired and checked by reading, not compiled, not run on hardware
//
#include <hip/hip_runtime.h>
#include <stdint.h>
#include <math.h>

#define H 512
#define LDEC 2
#define B 64
#define T 2048
#define K 128
#define R 7

typedef __attribute__((ext_vector_type(16))) __bf16         bf16x16;
typedef __attribute__((ext_vector_type(16))) unsigned short u16x16;
typedef __attribute__((ext_vector_type(8)))  unsigned int   u32x8;
typedef __attribute__((ext_vector_type(8)))  float          v8f;

__device__ __forceinline__ unsigned short f2bf(float f) {
    // round-to-nearest-even fp32 -> bf16
    unsigned int u = __float_as_uint(f);
    unsigned int r = u + 0x7FFFu + ((u >> 16) & 1u);
    return (unsigned short)(r >> 16);
}

// ---------------------------------------------------------------------------
// 1) Transpose + convert weights to bf16.
//    WeT[n*H + k] = bf16(We[k][n])   (n = output col, k contiguous)
//    WpT[n*K + k] = bf16(Wp[k][n])
// ---------------------------------------------------------------------------
__global__ void prep_weights(const float* __restrict__ We,
                             const float* __restrict__ Wp,
                             unsigned short* __restrict__ WeT,
                             unsigned short* __restrict__ WpT) {
    int idx = blockIdx.x * blockDim.x + threadIdx.x;
    if (idx < H * H) {
        int n = idx / H, k = idx % H;
        WeT[n * H + k] = f2bf(We[k * H + n]);
    } else {
        int j = idx - H * H;
        if (j < H * K) {
            int n = j / K, k = j % K;
            WpT[n * K + k] = f2bf(Wp[k * H + n]);
        }
    }
}

// ---------------------------------------------------------------------------
// 2) base[b][n] = (mean_l hidden[l][b][:]) @ Wh + bh + be + bp   (fp32)
// ---------------------------------------------------------------------------
__global__ void prep_base(const float* __restrict__ hidden,
                          const float* __restrict__ Wh,
                          const float* __restrict__ bh,
                          const float* __restrict__ be,
                          const float* __restrict__ bp,
                          float* __restrict__ base) {
    int idx = blockIdx.x * blockDim.x + threadIdx.x;
    if (idx >= B * H) return;
    int b = idx / H, n = idx % H;
    float acc = 0.f;
    for (int k = 0; k < H; ++k) {
        float hm = 0.f;
        #pragma unroll
        for (int l = 0; l < LDEC; ++l)
            hm += hidden[(size_t)l * B * H + (size_t)b * H + k];
        acc += (hm * (1.0f / (float)LDEC)) * Wh[(size_t)k * H + n];
    }
    base[idx] = acc + bh[n] + be[n] + bp[n];
}

// ---------------------------------------------------------------------------
// 3) conv1d(1->K, width 7, pad 3) over prev_attention, store bf16 (B,T,K)
// ---------------------------------------------------------------------------
__global__ void conv_kernel(const float* __restrict__ prev,
                            const float* __restrict__ conv_w,
                            const float* __restrict__ conv_b,
                            unsigned short* __restrict__ convA) {
    int idx = blockIdx.x * blockDim.x + threadIdx.x;  // (b,t,k), k fastest
    int k = idx % K;
    int t = (idx / K) % T;
    int b = idx / (K * T);
    float acc = conv_b[k];
    #pragma unroll
    for (int r = 0; r < R; ++r) {
        int tt = t + r - 3;
        float x = (tt >= 0 && tt < T) ? prev[(size_t)b * T + tt] : 0.f;
        acc += x * conv_w[k * R + r];
    }
    convA[idx] = f2bf(acc);
}

// ---------------------------------------------------------------------------
// 4) Fused score kernel (the WMMA workhorse).
//    Per wave: 16 t-rows of one batch b. The wave's full 16x512 encoder
//    A-slab is converted to bf16 fragments ONCE and kept resident in 128
//    VGPRs. Then loop over all 32 column tiles: 16 enc-k + 4 conv-k
//    v_wmma_f32_16x16x32_bf16 each, streaming only B (L2-resident weights)
//    and the wave-private bf16 conv fragments. Epilogue: tanh + Wo dot,
//    shuffle-reduce columns -> energy(B,T).
//
//    A layout (ISA 7.12.2, 16-bit A 16x32): lane%16 = M; per-lane K runs:
//      [kb + 8*half .. +7] then [kb+16+8*half .. +7]  (two 8-elem runs)
//    B layout: lane%16 = N; per-lane K runs same as A: [kb + 8*half .. +7] then [kb+16+8*half .. +7].
//    C layout: VGPR r: lanes0-15 -> M=r, lanes16-31 -> M=8+r; N = lane%16.
// ---------------------------------------------------------------------------
__global__ void __launch_bounds__(256)
score_wmma(const float* __restrict__ enc,              // (T,B,H) fp32
           const unsigned short* __restrict__ WeT,     // (H,H) bf16
           const unsigned short* __restrict__ WpT,     // (H,K) bf16
           const unsigned short* __restrict__ convA,   // (B,T,K) bf16
           const float* __restrict__ base,             // (B,H)
           const float* __restrict__ Wo,
           const float* __restrict__ bo,
           float* __restrict__ energy) {               // (B,T)
    const int lane = threadIdx.x & 31;
    const int wave = threadIdx.x >> 5;
    const int b    = blockIdx.x / (T / 128);
    const int tblk = blockIdx.x % (T / 128);
    const int t0   = tblk * 128 + wave * 16;
    const int m    = lane & 15;
    const int half = lane >> 4;

    const int trow = t0 + m;
    const float*          encRow  = enc   + ((size_t)trow * B + b) * H;
    const unsigned short* convRow = convA + ((size_t)b * T + trow) * K;

    // ---- stage the 16x512 encoder A slab in registers (converted once) ----
    bf16x16 aslab[16];
    #pragma unroll
    for (int kt = 0; kt < 16; ++kt) {
        const int kb = kt * 32;
        const float4* p0 = (const float4*)(encRow + kb + 8 * half);
        const float4* p1 = (const float4*)(encRow + kb + 16 + 8 * half);
        float4 a0 = p0[0], a1 = p0[1];
        float4 a2 = p1[0], a3 = p1[1];
        u16x16 au;
        au[0]  = f2bf(a0.x); au[1]  = f2bf(a0.y);
        au[2]  = f2bf(a0.z); au[3]  = f2bf(a0.w);
        au[4]  = f2bf(a1.x); au[5]  = f2bf(a1.y);
        au[6]  = f2bf(a1.z); au[7]  = f2bf(a1.w);
        au[8]  = f2bf(a2.x); au[9]  = f2bf(a2.y);
        au[10] = f2bf(a2.z); au[11] = f2bf(a2.w);
        au[12] = f2bf(a3.x); au[13] = f2bf(a3.y);
        au[14] = f2bf(a3.z); au[15] = f2bf(a3.w);
        aslab[kt] = __builtin_bit_cast(bf16x16, au);
    }

    float ep[8];
    #pragma unroll
    for (int r = 0; r < 8; ++r) ep[r] = 0.f;

    const float bsH = 0.f;  // (kept for clarity; bias fetched per tile below)
    (void)bsH;

    for (int j = 0; j < 32; ++j) {               // 32 column tiles of 16
        const int n = j * 16 + m;
        v8f acc;
        #pragma unroll
        for (int r = 0; r < 8; ++r) acc[r] = 0.f;

        // ---- encoder path: 16 k-tiles of 32 (A resident in registers) ----
        const unsigned short* brow = WeT + (size_t)n * H;
        #pragma unroll
        for (int kt = 0; kt < 16; ++kt) {
            uint4 q0 = *(const uint4*)(brow + kt * 32 + 8 * half);
            uint4 q1 = *(const uint4*)(brow + kt * 32 + 16 + 8 * half);
            u32x8 bw = {q0.x, q0.y, q0.z, q0.w, q1.x, q1.y, q1.z, q1.w};
            bf16x16 Bv = __builtin_bit_cast(bf16x16, bw);
            acc = __builtin_amdgcn_wmma_f32_16x16x32_bf16(
                false, aslab[kt], false, Bv, (short)0, acc, false, false);
        }
        // ---- conv path: 4 k-tiles of 32 (bf16 fragments, L0-cached) ----
        const unsigned short* prow = WpT + (size_t)n * K;
        #pragma unroll
        for (int kt = 0; kt < 4; ++kt) {
            const int kb = kt * 32;
            uint4 c0 = *(const uint4*)(convRow + kb + 8 * half);
            uint4 c1 = *(const uint4*)(convRow + kb + 16 + 8 * half);
            u32x8 aw = {c0.x, c0.y, c0.z, c0.w, c1.x, c1.y, c1.z, c1.w};
            bf16x16 Av = __builtin_bit_cast(bf16x16, aw);

            uint4 q0 = *(const uint4*)(prow + kb + 8 * half);
            uint4 q1 = *(const uint4*)(prow + kb + 16 + 8 * half);
            u32x8 bw = {q0.x, q0.y, q0.z, q0.w, q1.x, q1.y, q1.z, q1.w};
            bf16x16 Bv = __builtin_bit_cast(bf16x16, bw);

            acc = __builtin_amdgcn_wmma_f32_16x16x32_bf16(
                false, Av, false, Bv, (short)0, acc, false, false);
        }
        // ---- epilogue: tanh(acc + base[col]) * Wo[col], accumulate per row ----
        float bias = base[(size_t)b * H + n];
        float wo   = Wo[n];
        #pragma unroll
        for (int r = 0; r < 8; ++r)
            ep[r] += tanhf(acc[r] + bias) * wo;
    }

    // reduce over the 16 columns held by each half-wave
    #pragma unroll
    for (int r = 0; r < 8; ++r) {
        float v = ep[r];
        v += __shfl_xor(v, 1, 32);
        v += __shfl_xor(v, 2, 32);
        v += __shfl_xor(v, 4, 32);
        v += __shfl_xor(v, 8, 32);
        ep[r] = v;
    }
    if (m == 0) {
        float b0 = bo[0];
        #pragma unroll
        for (int r = 0; r < 8; ++r) {
            int row = t0 + half * 8 + r;   // C layout: VGPR r -> M = r + 8*half
            energy[(size_t)b * T + row] = ep[r] + b0;
        }
    }
}

// ---------------------------------------------------------------------------
// 5) Length-masked softmax over T per batch row. out = (B,T,1) fp32.
// ---------------------------------------------------------------------------
__global__ void softmax_kernel(const float* __restrict__ energy,
                               const int* __restrict__ enc_len,
                               float* __restrict__ out) {
    __shared__ float red[256];
    int b   = blockIdx.x;
    int tid = threadIdx.x;
    int len = enc_len[b];
    const float* e = energy + (size_t)b * T;
    float*       o = out    + (size_t)b * T;

    float mx = -3.0e38f;
    for (int t = tid; t < T; t += 256)
        if (t < len) mx = fmaxf(mx, e[t]);
    red[tid] = mx; __syncthreads();
    for (int s = 128; s > 0; s >>= 1) {
        if (tid < s) red[tid] = fmaxf(red[tid], red[tid + s]);
        __syncthreads();
    }
    mx = red[0]; __syncthreads();

    float sum = 0.f;
    for (int t = tid; t < T; t += 256) {
        float p = (t < len) ? expf(e[t] - mx) : 0.f;
        o[t] = p;
        sum += p;
    }
    red[tid] = sum; __syncthreads();
    for (int s = 128; s > 0; s >>= 1) {
        if (tid < s) red[tid] += red[tid + s];
        __syncthreads();
    }
    float inv = 1.0f / red[0];
    for (int t = tid; t < T; t += 256)
        o[t] *= inv;
}

// ---------------------------------------------------------------------------
extern "C" void kernel_launch(void* const* d_in, const int* in_sizes, int n_in,
                              void* d_out, int out_size, void* d_ws, size_t ws_size,
                              hipStream_t stream) {
    const float* hidden  = (const float*)d_in[0];
    const float* enc     = (const float*)d_in[1];
    const int*   enc_len = (const int*)  d_in[2];
    const float* prev    = (const float*)d_in[3];
    const float* Wh      = (const float*)d_in[4];
    const float* bh      = (const float*)d_in[5];
    const float* We      = (const float*)d_in[6];
    const float* be      = (const float*)d_in[7];
    const float* conv_w  = (const float*)d_in[8];
    const float* conv_b  = (const float*)d_in[9];
    const float* Wp      = (const float*)d_in[10];
    const float* bp      = (const float*)d_in[11];
    const float* Wo      = (const float*)d_in[12];
    const float* bo      = (const float*)d_in[13];
    float* out = (float*)d_out;

    char* ws = (char*)d_ws;
    unsigned short* WeT    = (unsigned short*)ws; ws += (size_t)H * H * 2;      // 512 KB
    unsigned short* WpT    = (unsigned short*)ws; ws += (size_t)H * K * 2;      // 128 KB
    float*          base   = (float*)ws;          ws += (size_t)B * H * 4;      // 128 KB
    unsigned short* convA  = (unsigned short*)ws; ws += (size_t)B * T * K * 2;  //  32 MB
    float*          energy = (float*)ws;                                         // 512 KB

    prep_weights<<<(H * H + H * K + 255) / 256, 256, 0, stream>>>(We, Wp, WeT, WpT);
    prep_base   <<<(B * H + 255) / 256,        256, 0, stream>>>(hidden, Wh, bh, be, bp, base);
    conv_kernel <<<(B * T * K) / 256,          256, 0, stream>>>(prev, conv_w, conv_b, convA);
    score_wmma  <<<B * (T / 128),              256, 0, stream>>>(enc, WeT, WpT, convA, base,
                                                                 Wo, bo, energy);
    softmax_kernel<<<B, 256, 0, stream>>>(energy, enc_len, out);
}
